// MultiHeadAttention_45715631898862
// MI455X (gfx1250) — matrix operand order repaired and checked by reading, not compiled, not run on hardware
//
#include <hip/hip_runtime.h>

// MHA: B=2, T=2048, C=1024, H=16, D=64.  bf16 WMMA compute, fp32 accumulate/softmax.

#define Bsz  2
#define Tseq 2048
#define Cdim 1024
#define Hn   16
#define Dh   64
#define BT   (Bsz * Tseq)   // 4096

typedef __bf16 v16bf __attribute__((ext_vector_type(16)));
typedef __bf16 v8bf  __attribute__((ext_vector_type(8)));
typedef __bf16 v4bf  __attribute__((ext_vector_type(4)));
typedef float  v8f   __attribute__((ext_vector_type(8)));

// Per-lane async 16B copy global -> LDS (ASYNCcnt-tracked). LDS dest address and
// 64-bit global address live in VGPRs; low 32 bits of a flat shared pointer are the
// wave-relative LDS byte offset (ISA 10.2 aperture mapping).
__device__ __forceinline__ void async_copy16(const void* g, void* lds) {
  unsigned ldsb = (unsigned)(size_t)lds;
  unsigned long long ga = (unsigned long long)(size_t)g;
  asm volatile("global_load_async_to_lds_b128 %0, %1, off" :: "v"(ldsb), "v"(ga) : "memory");
}

// ---------------------------------------------------------------- cast f32 -> bf16
__global__ __launch_bounds__(256) void cast_f32_to_bf16(const float* __restrict__ src,
                                                        __bf16* __restrict__ dst, int n4) {
  int i = blockIdx.x * 256 + threadIdx.x;
  if (i < n4) {
    float4 f = reinterpret_cast<const float4*>(src)[i];
    v4bf o;
    o[0] = (__bf16)f.x; o[1] = (__bf16)f.y; o[2] = (__bf16)f.z; o[3] = (__bf16)f.w;
    reinterpret_cast<v4bf*>(dst)[i] = o;
  }
}

// ---------------------------------------------------------------- GEMM: Out = A @ W^T + bias
// A: [M=4096, K=1024] bf16 row-major, W: [N=1024, K=1024] bf16 row-major (nn.Linear layout).
// One wave -> 32x64 output tile (2 A frags x 4 B frags = 8 WMMA accumulators): each B
// fragment feeds two WMMAs. Waves of a block cover consecutive 32-row tiles of the SAME
// 64-col W panel, so B loads are identical across waves (L0/WGP$ hits).
// mode 0: bf16 out [B,H,T,D] * scale   (Q / K heads)
// mode 1: bf16 out [B,H,D,T]           (V transposed for P*V B-fragments)
// mode 2: f32  out [BT, C]             (final projection)
__global__ __launch_bounds__(256) void gemm_bf16_nt(const __bf16* __restrict__ A,
                                                    const __bf16* __restrict__ W,
                                                    const float*  __restrict__ bias,
                                                    void* __restrict__ outp,
                                                    int mode, float scale) {
  const int K = Cdim;
  int lane = threadIdx.x & 31;
  int wid  = blockIdx.x * 8 + (threadIdx.x >> 5);   // 2048 wave tiles
  int mt = wid & 127;                               // 128 tiles of 32 rows
  int nt = wid >> 7;                                // 16 panels of 64 cols
  int m0 = mt << 5;
  int n0 = nt << 6;

  int half = lane >> 4;          // 0|1
  int l15  = lane & 15;
  int akb  = half << 4;          // A frag K sub-offset: 0|16
  int bkb  = half << 4;          // B frag K sub-offset: 0|16

  v8f acc[2][4];
#pragma unroll
  for (int g = 0; g < 2; ++g)
#pragma unroll
    for (int t = 0; t < 4; ++t)
#pragma unroll
      for (int r = 0; r < 8; ++r) acc[g][t][r] = 0.f;

  const __bf16* arow0 = A + (size_t)(m0 + l15) * K;
  const __bf16* arow1 = A + (size_t)(m0 + 16 + l15) * K;

  for (int k0 = 0; k0 < K; k0 += 32) {
    v16bf a0, a1;
    {
      const v8bf* p0 = reinterpret_cast<const v8bf*>(arow0 + k0 + akb);
      const v8bf* p1 = reinterpret_cast<const v8bf*>(arow1 + k0 + akb);
      v8bf lo0 = p0[0], hi0 = p0[1];
      v8bf lo1 = p1[0], hi1 = p1[1];
#pragma unroll
      for (int i = 0; i < 8; ++i) {
        a0[i] = lo0[i]; a0[i + 8] = hi0[i];
        a1[i] = lo1[i]; a1[i + 8] = hi1[i];
      }
    }
#pragma unroll
    for (int t = 0; t < 4; ++t) {
      int n = n0 + t * 16 + l15;
      v16bf b = *reinterpret_cast<const v16bf*>(W + (size_t)n * K + k0 + bkb);
      acc[0][t] = __builtin_amdgcn_wmma_f32_16x16x32_bf16(false, a0, false, b,
                                                          (short)0, acc[0][t], false, false);
      acc[1][t] = __builtin_amdgcn_wmma_f32_16x16x32_bf16(false, a1, false, b,
                                                          (short)0, acc[1][t], false, false);
    }
  }

#pragma unroll
  for (int g = 0; g < 2; ++g) {
#pragma unroll
    for (int t = 0; t < 4; ++t) {
#pragma unroll
      for (int r = 0; r < 8; ++r) {
        int row = m0 + g * 16 + r + (half << 3);
        int col = n0 + t * 16 + l15;
        float v = acc[g][t][r] + bias[col];
        if (mode == 0) {                       // [B,H,T,D] bf16, scaled
          int b = row >> 11, tt = row & (Tseq - 1);
          int h = col >> 6,  d  = col & (Dh - 1);
          ((__bf16*)outp)[((((size_t)b * Hn + h) * Tseq + tt) << 6) + d] = (__bf16)(v * scale);
        } else if (mode == 1) {                // [B,H,D,T] bf16 (V^T)
          int b = row >> 11, tt = row & (Tseq - 1);
          int h = col >> 6,  d  = col & (Dh - 1);
          ((__bf16*)outp)[(((size_t)b * Hn + h) * Dh + d) * Tseq + tt] = (__bf16)v;
        } else {                               // [BT, C] f32
          ((float*)outp)[(size_t)row * Cdim + col] = v;
        }
      }
    }
  }
}

// ---------------------------------------------------------------- fused flash attention
// Qh/Kh: [B*H, T, D] bf16 (Q pre-scaled by 1/sqrt(D)), Vt: [B*H, D, T] bf16.
// One block (8 waves) = 256 query rows of ONE head; each wave owns 32 queries.
// All waves share the K/V stream: 32-key K (4KB) and V (4KB) tiles are staged in LDS by
// global_load_async_to_lds_b128, double-buffered (s_wait_asynccnt + barrier), so global
// K/V traffic drops 8x and the next tile streams in under the WMMAs.
__global__ __launch_bounds__(256) void attn_fused(const __bf16* __restrict__ Qh,
                                                  const __bf16* __restrict__ Kh,
                                                  const __bf16* __restrict__ Vt,
                                                  __bf16* __restrict__ Oc) {
  // [0,16K): K/V double buffer (2 x (4KB K + 4KB V));  [16K,32K): per-wave 32x32 P staging
  __shared__ __align__(16) char smem[16384 + 8 * 2048];

  int tid   = threadIdx.x;
  int lane  = tid & 31;
  int wslot = tid >> 5;
  int bh = blockIdx.x >> 3;                       // 32 heads, 8 blocks/head
  int q0 = ((blockIdx.x & 7) * 8 + wslot) * 32;   // 32 queries per wave

  const __bf16* Qbase = Qh + (size_t)bh * Tseq * Dh;
  const __bf16* Kbase = Kh + (size_t)bh * Tseq * Dh;
  const __bf16* Vbase = Vt + (size_t)bh * Dh * Tseq;

  int half = lane >> 4;
  int l15  = lane & 15;
  int akb  = half << 4;
  int bkb  = half << 4;

  // Q A-fragments: 2 row-groups (16 rows each) x 2 K=32 fragments, held in regs.
  v16bf aQ[2][2];
#pragma unroll
  for (int g = 0; g < 2; ++g)
#pragma unroll
    for (int f = 0; f < 2; ++f) {
      const v8bf* qp = reinterpret_cast<const v8bf*>(
          Qbase + (size_t)(q0 + g * 16 + l15) * Dh + f * 32 + akb);
      v8bf lo = qp[0], hi = qp[1];
#pragma unroll
      for (int i = 0; i < 8; ++i) { aQ[g][f][i] = lo[i]; aQ[g][f][i + 8] = hi[i]; }
    }

  float mrow[2][8], lrow[2][8];
  v8f accO[2][4];
#pragma unroll
  for (int g = 0; g < 2; ++g)
#pragma unroll
    for (int r = 0; r < 8; ++r) { mrow[g][r] = -3.0e30f; lrow[g][r] = 0.f; }
#pragma unroll
  for (int g = 0; g < 2; ++g)
#pragma unroll
    for (int t = 0; t < 4; ++t)
#pragma unroll
      for (int r = 0; r < 8; ++r) accO[g][t][r] = 0.f;

  __bf16* pl = (__bf16*)(smem + 16384 + wslot * 2048);

  // This thread's 16B copy slices: K tile = 32 rows x 128B, V tile = 64 rows x 64B.
  const __bf16* gKb = Kbase + (size_t)(tid >> 3) * Dh + (tid & 7) * 8;
  const __bf16* gVb = Vbase + (size_t)(tid >> 2) * Tseq + (tid & 3) * 8;

  const int nsteps = Tseq / 32;
  auto issue = [&](int s) {
    char* buf = smem + (s & 1) * 8192;
    async_copy16(gKb + (size_t)s * 32 * Dh, buf + tid * 16);          // K tile
    async_copy16(gVb + s * 32,              buf + 4096 + tid * 16);   // V tile
  };

  issue(0);
  for (int s = 0; s < nsteps; ++s) {
    asm volatile("s_wait_asynccnt 0" ::: "memory");   // own async stores landed in LDS
    __syncthreads();                                  // everyone's tile visible
    if (s + 1 < nsteps) issue(s + 1);                 // prefetch next buffer

    const __bf16* bK = (const __bf16*)(smem + (s & 1) * 8192);
    const __bf16* bV = (const __bf16*)(smem + (s & 1) * 8192 + 4096);

    // K^T B-fragments from LDS: [jj = key subtile][kf = K-chunk], each used by both groups.
    v16bf bKf[2][2];
#pragma unroll
    for (int jj = 0; jj < 2; ++jj)
#pragma unroll
      for (int kf = 0; kf < 2; ++kf)
        bKf[jj][kf] = *reinterpret_cast<const v16bf*>(
            bK + (size_t)(jj * 16 + l15) * Dh + kf * 32 + bkb);

    // ---- scores: S = Q * K^T (Q pre-scaled)
    v8f sc[2][2];
#pragma unroll
    for (int g = 0; g < 2; ++g)
#pragma unroll
      for (int jj = 0; jj < 2; ++jj) {
#pragma unroll
        for (int r = 0; r < 8; ++r) sc[g][jj][r] = 0.f;
#pragma unroll
        for (int kf = 0; kf < 2; ++kf)
          sc[g][jj] = __builtin_amdgcn_wmma_f32_16x16x32_bf16(
              false, aQ[g][kf], false, bKf[jj][kf], (short)0, sc[g][jj], false, false);
      }

    // ---- online softmax (row stats replicated across each 16-lane half)
#pragma unroll
    for (int g = 0; g < 2; ++g) {
#pragma unroll
      for (int r = 0; r < 8; ++r) {
        float tm = fmaxf(sc[g][0][r], sc[g][1][r]);
#pragma unroll
        for (int xm = 1; xm < 16; xm <<= 1) tm = fmaxf(tm, __shfl_xor(tm, xm, 32));
        float mnew  = fmaxf(mrow[g][r], tm);
        float alpha = __expf(mrow[g][r] - mnew);
        float p0 = __expf(sc[g][0][r] - mnew);
        float p1 = __expf(sc[g][1][r] - mnew);
        float rs = p0 + p1;
#pragma unroll
        for (int xm = 1; xm < 16; xm <<= 1) rs += __shfl_xor(rs, xm, 32);
        lrow[g][r] = lrow[g][r] * alpha + rs;
        mrow[g][r] = mnew;
        sc[g][0][r] = p0; sc[g][1][r] = p1;
#pragma unroll
        for (int t = 0; t < 4; ++t) accO[g][t][r] *= alpha;
      }
    }

    // ---- transpose P (C-layout -> A-layout) through this wave's private LDS slot
#pragma unroll
    for (int g = 0; g < 2; ++g)
#pragma unroll
      for (int jj = 0; jj < 2; ++jj)
#pragma unroll
        for (int r = 0; r < 8; ++r)
          pl[g * 512 + (r + (half << 3)) * 32 + jj * 16 + l15] = (__bf16)sc[g][jj][r];

    asm volatile("s_wait_dscnt 0" ::: "memory");      // wave-internal LDS RAW ordering

    v16bf aP[2];
#pragma unroll
    for (int g = 0; g < 2; ++g) {
      const v8bf* pp = reinterpret_cast<const v8bf*>(pl + g * 512 + l15 * 32 + akb);
      v8bf lo = pp[0], hi = pp[1];
#pragma unroll
      for (int i = 0; i < 8; ++i) { aP[g][i] = lo[i]; aP[g][i + 8] = hi[i]; }
    }

    // ---- accO += P (32x32) * V (32x64); each V B-fragment feeds both row-groups
#pragma unroll
    for (int t = 0; t < 4; ++t) {
      v16bf bVf = *reinterpret_cast<const v16bf*>(bV + (size_t)(t * 16 + l15) * 32 + bkb);
      accO[0][t] = __builtin_amdgcn_wmma_f32_16x16x32_bf16(false, aP[0], false, bVf,
                                                           (short)0, accO[0][t], false, false);
      accO[1][t] = __builtin_amdgcn_wmma_f32_16x16x32_bf16(false, aP[1], false, bVf,
                                                           (short)0, accO[1][t], false, false);
    }
  }

  // ---- finalize: divide by row sums, write [B, T, h*64+d] bf16
  int b = bh >> 4, h = bh & 15;
#pragma unroll
  for (int g = 0; g < 2; ++g) {
#pragma unroll
    for (int r = 0; r < 8; ++r) {
      float inv = 1.0f / lrow[g][r];
      int t = q0 + g * 16 + r + (half << 3);
#pragma unroll
      for (int nt = 0; nt < 4; ++nt) {
        int col = (h << 6) + nt * 16 + l15;
        Oc[((size_t)b * Tseq + t) * Cdim + col] = (__bf16)(accO[g][nt][r] * inv);
      }
    }
  }
}

// ---------------------------------------------------------------- launch
extern "C" void kernel_launch(void* const* d_in, const int* in_sizes, int n_in,
                              void* d_out, int out_size, void* d_ws, size_t ws_size,
                              hipStream_t stream) {
  const float* q  = (const float*)d_in[0];
  const float* k  = (const float*)d_in[1];
  const float* v  = (const float*)d_in[2];
  const float* Wq = (const float*)d_in[3];
  const float* bq = (const float*)d_in[4];
  const float* Wk = (const float*)d_in[5];
  const float* bk = (const float*)d_in[6];
  const float* Wv = (const float*)d_in[7];
  const float* bv = (const float*)d_in[8];
  const float* Wo = (const float*)d_in[9];
  const float* bo = (const float*)d_in[10];

  char* w = (char*)d_ws;
  const size_t MB = (size_t)1 << 20;
  __bf16* xq = (__bf16*)(w + 0 * MB);     // 4096x1024 bf16 activations
  __bf16* xk = (__bf16*)(w + 8 * MB);
  __bf16* xv = (__bf16*)(w + 16 * MB);
  __bf16* wq = (__bf16*)(w + 24 * MB);    // 1024x1024 bf16 weights
  __bf16* wk = (__bf16*)(w + 26 * MB);
  __bf16* wv = (__bf16*)(w + 28 * MB);
  __bf16* wo = (__bf16*)(w + 30 * MB);
  __bf16* qh = (__bf16*)(w + 32 * MB);    // [B,H,T,D]
  __bf16* kh = (__bf16*)(w + 40 * MB);    // [B,H,T,D]
  __bf16* vt = (__bf16*)(w + 48 * MB);    // [B,H,D,T]
  __bf16* ao = (__bf16*)(w + 56 * MB);    // [B,T,C] attention output

  const int NA = BT * Cdim;     // 4M elements
  const int NW = Cdim * Cdim;   // 1M elements
  cast_f32_to_bf16<<<NA / 4 / 256, 256, 0, stream>>>(q,  xq, NA / 4);
  cast_f32_to_bf16<<<NA / 4 / 256, 256, 0, stream>>>(k,  xk, NA / 4);
  cast_f32_to_bf16<<<NA / 4 / 256, 256, 0, stream>>>(v,  xv, NA / 4);
  cast_f32_to_bf16<<<NW / 4 / 256, 256, 0, stream>>>(Wq, wq, NW / 4);
  cast_f32_to_bf16<<<NW / 4 / 256, 256, 0, stream>>>(Wk, wk, NW / 4);
  cast_f32_to_bf16<<<NW / 4 / 256, 256, 0, stream>>>(Wv, wv, NW / 4);
  cast_f32_to_bf16<<<NW / 4 / 256, 256, 0, stream>>>(Wo, wo, NW / 4);

  // Q projection folds the 1/sqrt(D)=0.125 softmax scale (exact power of two in bf16).
  gemm_bf16_nt<<<256, 256, 0, stream>>>(xq, wq, bq, qh, 0, 0.125f);
  gemm_bf16_nt<<<256, 256, 0, stream>>>(xk, wk, bk, kh, 0, 1.0f);
  gemm_bf16_nt<<<256, 256, 0, stream>>>(xv, wv, bv, vt, 1, 1.0f);

  attn_fused<<<256, 256, 0, stream>>>(qh, kh, vt, ao);

  gemm_bf16_nt<<<256, 256, 0, stream>>>(ao, wo, bo, d_out, 2, 1.0f);
}
